// Model_18923625906932
// MI455X (gfx1250) — matrix operand order repaired and checked by reading, not compiled, not run on hardware
//
#include <hip/hip_runtime.h>

// ---------------------------------------------------------------------------
// MI455X / gfx1250 implementation. wave32, WMMA bf16 16x16x32 w/ f32 accum.
// Weights pre-swizzled to WMMA-fragment order (32B contiguous per lane).
// ---------------------------------------------------------------------------

typedef __attribute__((ext_vector_type(16))) __bf16 v16bf;
typedef __attribute__((ext_vector_type(8)))  __bf16 v8bf;
typedef __attribute__((ext_vector_type(8)))  float  v8f;

static constexpr int Bq  = 1024;
static constexpr int Nn  = 65536;
static constexpr int Fn  = 64;
static constexpr int Dd  = 512;
static constexpr int DBn = 1024;
static constexpr int Cn  = 96;
static constexpr int CHUNK = 128;   // query rows per distance/topk chunk

static __device__ __forceinline__ v8f wmma_bf16(v16bf a, v16bf b, v8f c) {
  return __builtin_amdgcn_wmma_f32_16x16x32_bf16(false, a, false, b, (short)0, c,
                                                 false, false);
}

// A fragment: 16xK tile (row-major, leading dim ld). Order used here (not the ISA table): lane&15 = M, lanes
// 16-31 carry K+16; elements are 16 contiguous K (K+0..7, K+8..15)
// -> two 16B vector loads.
static __device__ __forceinline__ v16bf load_a_bf16(const __bf16* src, int ld,
                                                    int kbase, int lane) {
  const __bf16* p = src + (size_t)(lane & 15) * ld + kbase + ((lane & 16) ? 16 : 0);
  v8bf lo = *(const v8bf*)p;
  v8bf hi = *(const v8bf*)(p + 8);
  v16bf a;
#pragma unroll
  for (int e = 0; e < 8; ++e) { a[e] = lo[e]; a[e + 8] = hi[e]; }
  return a;
}
static __device__ __forceinline__ v16bf load_a_f32(const float* src, int ld,
                                                   int kbase, int lane) {
  const float* p = src + (size_t)(lane & 15) * ld + kbase + ((lane & 16) ? 16 : 0);
  v8f lo = *(const v8f*)p;
  v8f hi = *(const v8f*)(p + 8);
  v16bf a;
#pragma unroll
  for (int e = 0; e < 8; ++e) { a[e] = (__bf16)lo[e]; a[e + 8] = (__bf16)hi[e]; }
  return a;
}
// B fragment from pre-swizzled weights: lane's 16 elements contiguous (32B).
static __device__ __forceinline__ v16bf load_b_packed(const __bf16* Wp, int ntiles,
                                                      int kt, int nt, int lane) {
  return *(const v16bf*)(Wp + (((size_t)kt * ntiles + nt) * 32 + lane) * 16);
}
// B fragment from a matrix stored as its transpose [N][K] row-major
// (cand_k [N][D] as the B side of q @ cand_k^T): contiguous 32B per lane.
static __device__ __forceinline__ v16bf load_b_nk(const __bf16* Wt, int ldk,
                                                  int kbase, int nbase, int lane) {
  return *(const v16bf*)(Wt + (size_t)(nbase + (lane & 15)) * ldk
                            + kbase + ((lane & 16) ? 16 : 0));
}

// ---------------------------------------------------------------------------
// Pack fp32 weight [K][N] into bf16 WMMA B-fragment order:
//   Wp[((kt*ntiles + nt)*32 + lane)*16 + e] = W[kt*32 + eK(lane,e)][nt*16 + (lane&15)]
// ---------------------------------------------------------------------------
__global__ void pack_b_kernel(const float* __restrict__ W,
                              __bf16* __restrict__ Wp, int K, int Nc) {
  int t = blockIdx.x * 256 + threadIdx.x;
  int ntiles = Nc >> 4;
  int total = (K >> 5) * ntiles * 32;
  if (t >= total) return;
  int lane = t & 31;
  int tile = t >> 5;
  int kt = tile / ntiles, nt = tile % ntiles;
  int n  = nt * 16 + (lane & 15);
  int kb = kt * 32 + ((lane & 16) ? 16 : 0);
  __bf16* dst = Wp + ((size_t)tile * 32 + lane) * 16;
#pragma unroll
  for (int e = 0; e < 16; ++e) dst[e] = (__bf16)W[(size_t)(kb + e) * Nc + n];
}

// ---------------------------------------------------------------------------
// Fused encoder: one block = 16 rows.
//   h  = z @ linW + lin_b
//   h += relu(h @ eW1 + e_b1) @ eW2 + e_b2
//   k  = LN(h)*mix_g + mix_b @ KW + K_b
// Outputs: k (bf16 always; fp32 for query path), ||k||^2, h (query path).
// ---------------------------------------------------------------------------
__global__ __launch_bounds__(256) void encode_kernel(
    const float* __restrict__ src,            // [rows, Fn]
    const __bf16* __restrict__ lin16, const float* __restrict__ lin_b,
    const __bf16* __restrict__ w1, const float* __restrict__ b1,
    const __bf16* __restrict__ w2, const float* __restrict__ b2,
    const float* __restrict__ mixg, const float* __restrict__ mixb,
    const __bf16* __restrict__ kw, const float* __restrict__ kb,
    __bf16* __restrict__ outK16, float* __restrict__ outSq,
    float* __restrict__ outK32, float* __restrict__ outH) {
  __shared__ char lds[64 * 1024];
  float*  hF  = (float*)lds;                       // 16x512 f32   (32KB)
  __bf16* tB  = (__bf16*)(lds + 32 * 1024);        // 16x1024 bf16 (32KB)
  float*  red = (float*)(lds + 48 * 1024);         // scratch in tail of tB

  const int lane = threadIdx.x & 31;
  const int wv   = threadIdx.x >> 5;
  const int row0 = blockIdx.x * 16;
  const int m0   = (lane & 16) ? 8 : 0;
  const int nl   = lane & 15;

  // ---- stage A: h = z @ linW + b  (16x512, K = 64) : 4 tiles/wave
  {
    v8f acc[4];
#pragma unroll
    for (int i = 0; i < 4; ++i) {
      float bias = lin_b[(wv + 8 * i) * 16 + nl];
#pragma unroll
      for (int r = 0; r < 8; ++r) acc[i][r] = bias;
    }
#pragma unroll
    for (int kt = 0; kt < Fn / 32; ++kt) {
      v16bf a = load_a_f32(src + (size_t)row0 * Fn, Fn, kt * 32, lane);
#pragma unroll
      for (int i = 0; i < 4; ++i)
        acc[i] = wmma_bf16(a, load_b_packed(lin16, Dd / 16, kt, wv + 8 * i, lane),
                           acc[i]);
    }
#pragma unroll
    for (int i = 0; i < 4; ++i) {
      int nb = (wv + 8 * i) * 16;
#pragma unroll
      for (int r = 0; r < 8; ++r) hF[(m0 + r) * Dd + nb + nl] = acc[i][r];
    }
  }
  __syncthreads();

  // ---- stage B: t = relu(h @ eW1 + b1)  (16x1024, K = 512) : 8 tiles/wave
  {
    v8f acc[8];
#pragma unroll
    for (int i = 0; i < 8; ++i) {
      float bias = b1[(wv + 8 * i) * 16 + nl];
#pragma unroll
      for (int r = 0; r < 8; ++r) acc[i][r] = bias;
    }
    for (int kt = 0; kt < Dd / 32; ++kt) {
      v16bf a = load_a_f32(hF, Dd, kt * 32, lane);
#pragma unroll
      for (int i = 0; i < 8; ++i)
        acc[i] = wmma_bf16(a, load_b_packed(w1, DBn / 16, kt, wv + 8 * i, lane),
                           acc[i]);
    }
#pragma unroll
    for (int i = 0; i < 8; ++i) {
      int nb = (wv + 8 * i) * 16;
#pragma unroll
      for (int r = 0; r < 8; ++r)
        tB[(m0 + r) * DBn + nb + nl] = (__bf16)fmaxf(acc[i][r], 0.f);
    }
  }
  __syncthreads();

  // ---- stage C: h += t @ eW2 + b2  (16x512, K = 1024) : 4 tiles/wave
  {
    v8f acc[4];
#pragma unroll
    for (int i = 0; i < 4; ++i) {
      float bias = b2[(wv + 8 * i) * 16 + nl];
#pragma unroll
      for (int r = 0; r < 8; ++r) acc[i][r] = bias;
    }
    for (int kt = 0; kt < DBn / 32; ++kt) {
      v16bf a = load_a_bf16(tB, DBn, kt * 32, lane);
#pragma unroll
      for (int i = 0; i < 4; ++i)
        acc[i] = wmma_bf16(a, load_b_packed(w2, Dd / 16, kt, wv + 8 * i, lane),
                           acc[i]);
    }
#pragma unroll
    for (int i = 0; i < 4; ++i) {
      int nb = (wv + 8 * i) * 16;
#pragma unroll
      for (int r = 0; r < 8; ++r) hF[(m0 + r) * Dd + nb + nl] += acc[i][r];
    }
  }
  __syncthreads();

  // ---- stage D: LayerNorm (per row over 512), emit h (query path)
  {
    int r = threadIdx.x >> 4, c0 = threadIdx.x & 15;
    float s = 0.f, s2 = 0.f;
    for (int j = 0; j < 32; ++j) {
      float v = hF[r * Dd + c0 + 16 * j];
      s += v; s2 += v * v;
      if (outH) outH[(size_t)(row0 + r) * Dd + c0 + 16 * j] = v;
    }
    red[threadIdx.x] = s;
    red[256 + threadIdx.x] = s2;
  }
  __syncthreads();
  if (threadIdx.x < 16) {
    int r = threadIdx.x;
    float s = 0.f, s2 = 0.f;
    for (int c = 0; c < 16; ++c) { s += red[r * 16 + c]; s2 += red[256 + r * 16 + c]; }
    float m = s * (1.f / Dd);
    float var = s2 * (1.f / Dd) - m * m;
    red[512 + r] = m;
    red[528 + r] = rsqrtf(var + 1e-5f);
  }
  __syncthreads();
  {
    int r = threadIdx.x >> 4, c0 = threadIdx.x & 15;
    float m = red[512 + r], inv = red[528 + r];
    __bf16* lnB = tB;  // first 16KB of tB reused for LN output
    for (int j = 0; j < 32; ++j) {
      int c = c0 + 16 * j;
      lnB[r * Dd + c] = (__bf16)((hF[r * Dd + c] - m) * inv * mixg[c] + mixb[c]);
    }
  }
  __syncthreads();

  // ---- stage E: k = ln @ KW + K_b  (16x512, K = 512) : 4 tiles/wave
  {
    v8f acc[4];
#pragma unroll
    for (int i = 0; i < 4; ++i) {
      float bias = kb[(wv + 8 * i) * 16 + nl];
#pragma unroll
      for (int r = 0; r < 8; ++r) acc[i][r] = bias;
    }
    for (int kt = 0; kt < Dd / 32; ++kt) {
      v16bf a = load_a_bf16(tB, Dd, kt * 32, lane);
#pragma unroll
      for (int i = 0; i < 4; ++i)
        acc[i] = wmma_bf16(a, load_b_packed(kw, Dd / 16, kt, wv + 8 * i, lane),
                           acc[i]);
    }
#pragma unroll
    for (int i = 0; i < 4; ++i) {
      int nb = (wv + 8 * i) * 16;
#pragma unroll
      for (int r = 0; r < 8; ++r) {
        int m = m0 + r, n = nb + nl;
        float v = acc[i][r];
        hF[m * Dd + n] = v;
        outK16[(size_t)(row0 + m) * Dd + n] = (__bf16)v;
        if (outK32) outK32[(size_t)(row0 + m) * Dd + n] = v;
      }
    }
  }
  __syncthreads();
  {
    int r = threadIdx.x >> 4, c0 = threadIdx.x & 15;
    float s = 0.f;
    for (int j = 0; j < 32; ++j) { float v = hF[r * Dd + c0 + 16 * j]; s += v * v; }
    red[threadIdx.x] = s;
  }
  __syncthreads();
  if (threadIdx.x < 16) {
    int r = threadIdx.x;
    float s = 0.f;
    for (int c = 0; c < 16; ++c) s += red[r * 16 + c];
    outSq[row0 + r] = s;
  }
}

// ---------------------------------------------------------------------------
// Distance GEMM: d2[q, n] = |q|^2 + |c|^2 - 2 q.c over a 128-query chunk.
// grid = (N/128, CHUNK/16); each wave owns one 16-col tile.
// ---------------------------------------------------------------------------
__global__ __launch_bounds__(256) void dist_kernel(
    const __bf16* __restrict__ qk16, const float* __restrict__ qsq,
    const __bf16* __restrict__ ck16, const float* __restrict__ csq,
    float* __restrict__ d2, int qbase) {
  const int lane = threadIdx.x & 31;
  const int wv   = threadIdx.x >> 5;
  const int n0   = blockIdx.x * 128 + wv * 16;
  const int q0   = qbase + blockIdx.y * 16;
  v8f acc;
#pragma unroll
  for (int r = 0; r < 8; ++r) acc[r] = 0.f;
  for (int kt = 0; kt < Dd / 32; ++kt) {
    __builtin_prefetch(ck16 + (size_t)(n0 + (lane & 15)) * Dd + kt * 32 + 64, 0, 1);
    v16bf a = load_a_bf16(qk16 + (size_t)q0 * Dd, Dd, kt * 32, lane);
    v16bf b = load_b_nk(ck16, Dd, kt * 32, n0, lane);
    acc = wmma_bf16(a, b, acc);
  }
  const int m0 = (lane & 16) ? 8 : 0;
  const int n  = n0 + (lane & 15);
  float cs = csq[n];
#pragma unroll
  for (int r = 0; r < 8; ++r) {
    int m = m0 + r;
    d2[(size_t)(blockIdx.y * 16 + m) * Nn + n] = qsq[q0 + m] + cs - 2.f * acc[r];
  }
}

// ---------------------------------------------------------------------------
// Per-row top-96 smallest d2 via 11-bit radix histogram + tie resolution.
// Emits idx and sims = -d2 (order irrelevant downstream: softmax + sum).
// ---------------------------------------------------------------------------
__global__ __launch_bounds__(256) void topk_kernel(
    const float* __restrict__ d2, int* __restrict__ idxOut,
    float* __restrict__ simsOut, int qbase) {
  __shared__ unsigned hist[2048];
  __shared__ int      tieIdx[1024];
  __shared__ unsigned tieKey[1024];
  __shared__ float    tieVal[1024];
  __shared__ unsigned bestK[256];
  __shared__ int      bestI[256];
  __shared__ int      sc[4];  // T, need, outCnt, tieCnt

  const int t = threadIdx.x;
  const int row = blockIdx.x;
  const float* drow = d2 + (size_t)row * Nn;
  const int grow = qbase + row;

  for (int i = t; i < 2048; i += 256) hist[i] = 0u;
  __syncthreads();
  for (int n = t; n < Nn; n += 256) {
    unsigned u = __float_as_uint(drow[n]);
    u ^= (u & 0x80000000u) ? 0xFFFFFFFFu : 0x80000000u;
    atomicAdd(&hist[u >> 21], 1u);
  }
  __syncthreads();
  if (t == 0) {
    unsigned cum = 0; int T = 2047; int need = Cn;
    for (int b = 0; b < 2048; ++b) {
      unsigned c = hist[b];
      if (cum + c >= (unsigned)Cn) { T = b; need = Cn - (int)cum; break; }
      cum += c;
    }
    sc[0] = T; sc[1] = need; sc[2] = 0; sc[3] = 0;
  }
  __syncthreads();
  const int T = sc[0], need = sc[1];
  for (int n = t; n < Nn; n += 256) {
    float v = drow[n];
    unsigned u = __float_as_uint(v);
    u ^= (u & 0x80000000u) ? 0xFFFFFFFFu : 0x80000000u;
    int b = (int)(u >> 21);
    if (b < T) {
      int p = atomicAdd(&sc[2], 1);
      idxOut[grow * Cn + p] = n;
      simsOut[grow * Cn + p] = -v;
    } else if (b == T) {
      int p = atomicAdd(&sc[3], 1);
      if (p < 1024) { tieIdx[p] = n; tieKey[p] = u; tieVal[p] = v; }
    }
  }
  __syncthreads();
  const int tn = sc[3] < 1024 ? sc[3] : 1024;
  const int base = sc[2];
  for (int j = 0; j < need; ++j) {
    unsigned bk = 0xFFFFFFFFu; int bi = -1;
    for (int p = t; p < tn; p += 256)
      if (tieKey[p] < bk) { bk = tieKey[p]; bi = p; }
    bestK[t] = bk; bestI[t] = bi;
    __syncthreads();
    if (t == 0) {
      unsigned mb = 0xFFFFFFFFu; int mi = -1;
      for (int q = 0; q < 256; ++q)
        if (bestK[q] < mb) { mb = bestK[q]; mi = bestI[q]; }
      if (mi >= 0) {
        idxOut[grow * Cn + base + j] = tieIdx[mi];
        simsOut[grow * Cn + base + j] = -tieVal[mi];
        tieKey[mi] = 0xFFFFFFFFu;
      } else {  // degenerate overflow fallback
        idxOut[grow * Cn + base + j] = tieIdx[0];
        simsOut[grow * Cn + base + j] = -tieVal[0];
      }
    }
    __syncthreads();
  }
}

// ---------------------------------------------------------------------------
// Context: per query b, softmax over sims, then
//   values = y_emb + relu((k - ctx_k) @ TW1 + Tb1) @ TW2
//   xo[b]  = x_h[b] + sum_c probs[c] * values[c]
// ---------------------------------------------------------------------------
__global__ __launch_bounds__(256) void context_kernel(
    const int* __restrict__ idx, const float* __restrict__ sims,
    const float* __restrict__ cand_y, const __bf16* __restrict__ ck16,
    const float* __restrict__ qk32, const float* __restrict__ xh,
    const __bf16* __restrict__ tw1, const float* __restrict__ tb1,
    const __bf16* __restrict__ tw2, const float* __restrict__ labW,
    const float* __restrict__ labB, float* __restrict__ xo) {
  __shared__ char lds[53248];
  __bf16* dB   = (__bf16*)lds;                     // 16x512 diff (16KB)
  __bf16* tB   = (__bf16*)(lds + 16 * 1024);       // 16x1024 (32KB)
  float*  ctxa = (float*)(lds + 48 * 1024);        // 512 f32
  float*  prob = (float*)(lds + 50 * 1024);        // 96
  float*  cy   = (float*)(lds + 50 * 1024 + 512);  // 96
  int*    cidx = (int*)(lds + 50 * 1024 + 1024);   // 96

  const int b = blockIdx.x;
  const int t = threadIdx.x;
  const int lane = t & 31, wv = t >> 5;
  const int m0 = (lane & 16) ? 8 : 0, nl = lane & 15;

  if (t < Cn) {
    int ci = idx[b * Cn + t];
    cidx[t] = ci;
    cy[t] = cand_y[ci];
    prob[t] = sims[b * Cn + t];
  }
  for (int i = t; i < Dd; i += 256) ctxa[i] = 0.f;
  __syncthreads();
  if (t == 0) {
    float mx = prob[0];
    for (int i = 1; i < Cn; ++i) mx = fmaxf(mx, prob[i]);
    float s = 0.f;
    for (int i = 0; i < Cn; ++i) { float e = __expf(prob[i] - mx); prob[i] = e; s += e; }
    float inv = 1.f / s;
    for (int i = 0; i < Cn; ++i) prob[i] *= inv;
  }
  __syncthreads();

  for (int ct = 0; ct < Cn / 16; ++ct) {
    {  // diff tile (bf16)
      int r = t >> 4, c0 = t & 15;
      int ci = cidx[ct * 16 + r];
      const float* q = qk32 + (size_t)b * Dd;
      const __bf16* ck = ck16 + (size_t)ci * Dd;
      for (int j = 0; j < 32; ++j) {
        int c = c0 + 16 * j;
        dB[r * Dd + c] = (__bf16)(q[c] - (float)ck[c]);
      }
    }
    __syncthreads();
    // GEMM1: relu(diff @ TW1 + Tb1) -> tB : 8 tiles/wave
    {
      v8f acc[8];
#pragma unroll
      for (int i = 0; i < 8; ++i) {
        float bias = tb1[(wv + 8 * i) * 16 + nl];
#pragma unroll
        for (int r = 0; r < 8; ++r) acc[i][r] = bias;
      }
      for (int kt = 0; kt < Dd / 32; ++kt) {
        v16bf a = load_a_bf16(dB, Dd, kt * 32, lane);
#pragma unroll
        for (int i = 0; i < 8; ++i)
          acc[i] = wmma_bf16(a, load_b_packed(tw1, DBn / 16, kt, wv + 8 * i, lane),
                             acc[i]);
      }
#pragma unroll
      for (int i = 0; i < 8; ++i) {
        int nb = (wv + 8 * i) * 16;
#pragma unroll
        for (int r = 0; r < 8; ++r)
          tB[(m0 + r) * DBn + nb + nl] = (__bf16)fmaxf(acc[i][r], 0.f);
      }
    }
    __syncthreads();
    // GEMM2 + y_emb + prob-weighted accumulate : 4 tiles/wave
    {
      v8f acc[4];
#pragma unroll
      for (int i = 0; i < 4; ++i)
#pragma unroll
        for (int r = 0; r < 8; ++r) acc[i][r] = 0.f;
      for (int kt = 0; kt < DBn / 32; ++kt) {
        v16bf a = load_a_bf16(tB, DBn, kt * 32, lane);
#pragma unroll
        for (int i = 0; i < 4; ++i)
          acc[i] = wmma_bf16(a, load_b_packed(tw2, Dd / 16, kt, wv + 8 * i, lane),
                             acc[i]);
      }
#pragma unroll
      for (int i = 0; i < 4; ++i) {
        int n = (wv + 8 * i) * 16 + nl;
        float lw = labW[n], lb = labB[n];
        float contrib = 0.f;
#pragma unroll
        for (int r = 0; r < 8; ++r) {
          int c = ct * 16 + m0 + r;
          contrib += prob[c] * (acc[i][r] + cy[c] * lw + lb);
        }
        atomicAdd(&ctxa[n], contrib);
      }
    }
    __syncthreads();
  }
  for (int i = t; i < Dd; i += 256)
    xo[(size_t)b * Dd + i] = xh[(size_t)b * Dd + i] + ctxa[i];
}

// ---------------------------------------------------------------------------
// Predictor block + head (2 GF total -> plain fp32 VALU, one block per query)
// ---------------------------------------------------------------------------
static __device__ __forceinline__ float block_sum(float v, float* red) {
  int t = threadIdx.x;
  red[t] = v;
  __syncthreads();
  for (int s = 128; s > 0; s >>= 1) {
    if (t < s) red[t] += red[t + s];
    __syncthreads();
  }
  float r = red[0];
  __syncthreads();
  return r;
}

__global__ __launch_bounds__(256) void head_kernel(
    const float* __restrict__ xo, const float* __restrict__ plng,
    const float* __restrict__ plnb, const float* __restrict__ pw1,
    const float* __restrict__ pb1, const float* __restrict__ pw2,
    const float* __restrict__ pb2, const float* __restrict__ hlng,
    const float* __restrict__ hlnb, const float* __restrict__ hw,
    const float* __restrict__ hbias, float* __restrict__ out) {
  __shared__ float xr[Dd], ln[Dd], t1[DBn], red[256];
  const int b = blockIdx.x, t = threadIdx.x;
  for (int i = t; i < Dd; i += 256) xr[i] = xo[(size_t)b * Dd + i];
  __syncthreads();

  float a0 = xr[t], a1 = xr[t + 256];
  float sum = block_sum(a0 + a1, red);
  float ssq = block_sum(a0 * a0 + a1 * a1, red);
  float m = sum * (1.f / Dd), inv = rsqrtf(ssq * (1.f / Dd) - m * m + 1e-5f);
  ln[t]       = (a0 - m) * inv * plng[t] + plnb[t];
  ln[t + 256] = (a1 - m) * inv * plng[t + 256] + plnb[t + 256];
  __syncthreads();

  for (int c = t; c < DBn; c += 256) {
    float acc = pb1[c];
    for (int k = 0; k < Dd; ++k) acc += ln[k] * pw1[(size_t)k * DBn + c];
    t1[c] = fmaxf(acc, 0.f);
  }
  __syncthreads();
  for (int c = t; c < Dd; c += 256) {
    float acc = pb2[c];
    for (int k = 0; k < DBn; ++k) acc += t1[k] * pw2[(size_t)k * Dd + c];
    xr[c] += acc;
  }
  __syncthreads();

  a0 = xr[t]; a1 = xr[t + 256];
  sum = block_sum(a0 + a1, red);
  ssq = block_sum(a0 * a0 + a1 * a1, red);
  m = sum * (1.f / Dd); inv = rsqrtf(ssq * (1.f / Dd) - m * m + 1e-5f);
  float r0 = fmaxf((a0 - m) * inv * hlng[t] + hlnb[t], 0.f);
  float r1 = fmaxf((a1 - m) * inv * hlng[t + 256] + hlnb[t + 256], 0.f);
  float p = block_sum(r0 * hw[t] + r1 * hw[t + 256], red);
  if (t == 0) out[b] = p + hbias[0];
}

// ---------------------------------------------------------------------------
// Launch
// ---------------------------------------------------------------------------
extern "C" void kernel_launch(void* const* d_in, const int* in_sizes, int n_in,
                              void* d_out, int out_size, void* d_ws, size_t ws_size,
                              hipStream_t stream) {
  (void)in_sizes; (void)n_in; (void)out_size; (void)ws_size;
  const float* x_num   = (const float*)d_in[0];
  const float* cnum    = (const float*)d_in[1];
  const float* cand_y  = (const float*)d_in[2];
  const float* lin_W   = (const float*)d_in[3];
  const float* lin_b   = (const float*)d_in[4];
  const float* e_W1    = (const float*)d_in[5];
  const float* e_b1    = (const float*)d_in[6];
  const float* e_W2    = (const float*)d_in[7];
  const float* e_b2    = (const float*)d_in[8];
  const float* mix_g   = (const float*)d_in[9];
  const float* mix_b   = (const float*)d_in[10];
  const float* K_W     = (const float*)d_in[11];
  const float* K_b     = (const float*)d_in[12];
  const float* lab_W   = (const float*)d_in[13];
  const float* lab_b   = (const float*)d_in[14];
  const float* T_W1    = (const float*)d_in[15];
  const float* T_b1    = (const float*)d_in[16];
  const float* T_W2    = (const float*)d_in[17];
  const float* p_ln_g  = (const float*)d_in[18];
  const float* p_ln_b  = (const float*)d_in[19];
  const float* p_W1    = (const float*)d_in[20];
  const float* p_b1    = (const float*)d_in[21];
  const float* p_W2    = (const float*)d_in[22];
  const float* p_b2    = (const float*)d_in[23];
  const float* h_ln_g  = (const float*)d_in[24];
  const float* h_ln_b  = (const float*)d_in[25];
  const float* h_W     = (const float*)d_in[26];
  const float* h_b     = (const float*)d_in[27];

  char* ws = (char*)d_ws;
  size_t off = 0;
  auto alloc = [&](size_t bytes) -> void* {
    void* p = ws + off;
    off = (off + bytes + 255) & ~(size_t)255;
    return p;
  };
  __bf16* lin16 = (__bf16*)alloc((size_t)Fn * Dd * 2);
  __bf16* w1_16 = (__bf16*)alloc((size_t)Dd * DBn * 2);
  __bf16* w2_16 = (__bf16*)alloc((size_t)DBn * Dd * 2);
  __bf16* kw16  = (__bf16*)alloc((size_t)Dd * Dd * 2);
  __bf16* tw1_16= (__bf16*)alloc((size_t)Dd * DBn * 2);
  __bf16* tw2_16= (__bf16*)alloc((size_t)DBn * Dd * 2);
  __bf16* ck16  = (__bf16*)alloc((size_t)Nn * Dd * 2);
  float*  csq   = (float*)alloc((size_t)Nn * 4);
  float*  qk32  = (float*)alloc((size_t)Bq * Dd * 4);
  __bf16* qk16  = (__bf16*)alloc((size_t)Bq * Dd * 2);
  float*  qsq   = (float*)alloc((size_t)Bq * 4);
  float*  xh    = (float*)alloc((size_t)Bq * Dd * 4);
  float*  d2    = (float*)alloc((size_t)CHUNK * Nn * 4);
  int*    idx   = (int*)alloc((size_t)Bq * Cn * 4);
  float*  sims  = (float*)alloc((size_t)Bq * Cn * 4);
  float*  xo    = (float*)alloc((size_t)Bq * Dd * 4);

  auto pack = [&](const float* s, __bf16* d, int K, int Nc) {
    int total = (K >> 5) * (Nc >> 4) * 32;
    pack_b_kernel<<<(total + 255) / 256, 256, 0, stream>>>(s, d, K, Nc);
  };
  pack(lin_W, lin16, Fn, Dd);
  pack(e_W1, w1_16, Dd, DBn);
  pack(e_W2, w2_16, DBn, Dd);
  pack(K_W, kw16, Dd, Dd);
  pack(T_W1, tw1_16, Dd, DBn);
  pack(T_W2, tw2_16, DBn, Dd);

  // Encode candidates (N/16 blocks) and queries (B/16 blocks).
  encode_kernel<<<Nn / 16, 256, 0, stream>>>(
      cnum, lin16, lin_b, w1_16, e_b1, w2_16, e_b2, mix_g, mix_b, kw16, K_b,
      ck16, csq, nullptr, nullptr);
  encode_kernel<<<Bq / 16, 256, 0, stream>>>(
      x_num, lin16, lin_b, w1_16, e_b1, w2_16, e_b2, mix_g, mix_b, kw16, K_b,
      qk16, qsq, qk32, xh);

  // Chunked distance + top-k (d2 scratch reused; stream order serializes).
  for (int c = 0; c < Bq / CHUNK; ++c) {
    int qbase = c * CHUNK;
    dist_kernel<<<dim3(Nn / 128, CHUNK / 16), 256, 0, stream>>>(
        qk16, qsq, ck16, csq, d2, qbase);
    topk_kernel<<<CHUNK, 256, 0, stream>>>(d2, idx, sims, qbase);
  }

  context_kernel<<<Bq, 256, 0, stream>>>(idx, sims, cand_y, ck16, qk32, xh,
                                         tw1_16, T_b1, tw2_16, lab_W, lab_b, xo);

  head_kernel<<<Bq, 256, 0, stream>>>(xo, p_ln_g, p_ln_b, p_W1, p_b1, p_W2,
                                      p_b2, h_ln_g, h_ln_b, h_W, h_b,
                                      (float*)d_out);
}
